// TPIEncoder_3925600109327
// MI455X (gfx1250) — matrix operand order repaired and checked by reading, not compiled, not run on hardware
//
#include <hip/hip_runtime.h>
#include <hip/hip_bf16.h>

// ---------------------------------------------------------------------------
// TPI encoder forward for gfx1250 (MI455X):
//   f16 WMMA GEMMs + flash attention, tiles DMA'd into LDS via the
//   Tensor Data Mover (tensor_load_to_lds, TENSORcnt), double-buffered.
// ---------------------------------------------------------------------------

typedef __attribute__((ext_vector_type(16))) _Float16 v16h;
typedef __attribute__((ext_vector_type(8)))  float    v8f;

#define DEVFN __device__ __forceinline__

static constexpr int Ldim = 4;
static constexpr int Edim = 512;
static constexpr int E3   = 1536;
static constexpr int Hh   = 8;
static constexpr int HD   = 64;
static constexpr int FFd  = 2048;
static constexpr int Sdim = 1024;
static constexpr int Bdim = 8;
static constexpr int Ntok = Sdim * Bdim;   // 8192

#ifndef __has_builtin
#define __has_builtin(x) 0
#endif
#if __has_builtin(__builtin_amdgcn_tensor_load_to_lds)
#define TPI_HAVE_TDM 1
#else
#define TPI_HAVE_TDM 0
#endif

DEVFN v8f wmma32(v16h a, v16h b, v8f c) {
  // D = A(16x32 f16) * B(32x16 f16) + C(16x16 f32)
  return __builtin_amdgcn_wmma_f32_16x16x32_f16(
      /*neg_a=*/false, a, /*neg_b=*/false, b,
      /*c_mod=*/(short)0, c, /*reuse_a=*/false, /*reuse_b=*/false);
}

// A-operand gather from an LDS tile stored [M][K] row-major.
// 16-bit A layout (matches B): lanes 0-15 hold K=0..15, lanes 16-31 K=16..31.
DEVFN v16h frag_a(const _Float16* t, int ld, int r, int hi) {
  v16h f; const int kb = hi * 16;
  #pragma unroll
  for (int i = 0; i < 16; ++i) f[i] = t[r * ld + kb + i];
  return f;
}

// B-operand gather from a tile stored [N][K] row-major (rows of W = cols of B).
// 16-bit B layout used here (matches A): lanes 0-15 hold K=0..15, lanes 16-31 K=16..31.
DEVFN v16h frag_b(const _Float16* t, int ld, int n, int hi) {
  v16h f; const int kb = hi * 16;
  #pragma unroll
  for (int i = 0; i < 16; ++i) f[i] = t[n * ld + kb + i];
  return f;
}

// B-operand gather from a tile stored [K][N] row-major (V tile in attention).
DEVFN v16h frag_bT(const _Float16* t, int ld, int n, int hi) {
  v16h f; const int kb = hi * 16;
  #pragma unroll
  for (int i = 0; i < 16; ++i) f[i] = t[(kb + i) * ld + n];
  return f;
}

// ---------------------------------------------------------------------------
// Tensor Data Mover: DMA a 2D f16 tile (tile_w elems x tile_h rows, global row
// stride row_stride elems) into LDS, with hardware row padding so LDS rows are
// (tile_w + pad) halves apart. pad_int: 3 => pad every 16 DW, 4 => every 32 DW.
// pad_amt: 3 => 4 DWORDs (8 halves) of pad.
// ---------------------------------------------------------------------------
#if TPI_HAVE_TDM
typedef __attribute__((ext_vector_type(4))) unsigned int tdm_u32x4;
typedef __attribute__((ext_vector_type(8))) int          tdm_i32x8;
typedef __attribute__((ext_vector_type(4))) int          tdm_i32x4;

DEVFN void tdm_load_2d(unsigned lds_addr, const void* gptr,
                       unsigned tile_w, unsigned tile_h,
                       unsigned row_stride, unsigned pad_int, unsigned pad_amt) {
  const unsigned long long ga = (unsigned long long)gptr;
  tdm_u32x4 g0;
  g0[0] = 1u;                                   // count=1 (valid), user mode
  g0[1] = lds_addr;                             // D#.lds_addr (bytes)
  g0[2] = (unsigned)ga;                         // global_addr[31:0]
  g0[3] = (unsigned)(ga >> 32) | (2u << 30);    // global_addr[56:32] | type=2
  tdm_i32x8 g1;
  g1[0] = (int)((1u << 16) |                    // data_size = 2 bytes
                (1u << 20) |                    // pad_enable
                (pad_int << 22) | (pad_amt << 25));
  g1[1] = (int)((row_stride & 0xFFFFu) << 16);                     // tensor_dim0[15:0]
  g1[2] = (int)((row_stride >> 16) | ((tile_h & 0xFFFFu) << 16));  // dim0 hi | dim1 lo
  g1[3] = (int)((tile_w & 0xFFFFu) << 16);                         // dim1 hi=0 | tile_dim0
  g1[4] = (int)(tile_h & 0xFFFFu);                                 // tile_dim1, tile_dim2=0
  g1[5] = (int)row_stride;                                         // dim0_stride[31:0]
  g1[6] = 0;                                                       // stride hi / dim1_stride
  g1[7] = 0;
  const tdm_i32x4 gz = {0, 0, 0, 0};
#if __clang_major__ >= 23
  const tdm_i32x8 z8 = {0, 0, 0, 0, 0, 0, 0, 0};
  __builtin_amdgcn_tensor_load_to_lds(g0, g1, gz, gz, z8, 0);
#else
  __builtin_amdgcn_tensor_load_to_lds(g0, g1, gz, gz, 0);
#endif
}
#endif  // TPI_HAVE_TDM

// ---------------------------------------------------------------------------
// Elementwise helpers
// ---------------------------------------------------------------------------
__global__ void tpi_cvt_f16_kernel(const float* __restrict__ in,
                                   _Float16* __restrict__ out, int n) {
  int i = blockIdx.x * blockDim.x + threadIdx.x;
  if (i < n) out[i] = (_Float16)in[i];
}

__global__ void tpi_copy_f32_kernel(const float* __restrict__ in,
                                    float* __restrict__ out, int n) {
  int i = blockIdx.x * blockDim.x + threadIdx.x;
  if (i < n) out[i] = in[i];
}

// ---------------------------------------------------------------------------
// GEMM: C[M,N] = A[M,K] * W[N,K]^T + bias[N]   (f16 in, f32 acc)
// Block tile 64x64, 128 threads = 4 waves; each wave owns 16 rows x 64 cols.
// K tiles arrive via TDM, double-buffered so WMMA overlaps the DMA.
// ---------------------------------------------------------------------------
template <bool RELU, bool OUTF32, bool OUTF16>
__global__ __launch_bounds__(128)
void tpi_gemm_kernel(const _Float16* __restrict__ A,
                     const _Float16* __restrict__ W,
                     const float* __restrict__ bias,
                     float* __restrict__ Cf,
                     _Float16* __restrict__ Ch,
                     int M, int N, int K) {
  __shared__ _Float16 As[2][64][40];   // 32 halves + 8 pad per row
  __shared__ _Float16 Bs[2][64][40];

  const int tid  = threadIdx.x;
  const int wave = tid >> 5;
  const int lane = tid & 31;
  const int r  = lane & 15;
  const int hi = lane >> 4;

  const int rowBase = blockIdx.y * 64;
  const int colBase = blockIdx.x * 64;

  const _Float16* Abase = A + (size_t)rowBase * K;
  const _Float16* Wbase = W + (size_t)colBase * K;

  v8f acc[4] = {};

#if TPI_HAVE_TDM
  if (wave == 0) {
    tdm_load_2d((unsigned)(size_t)&As[0][0][0], Abase, 32, 64, (unsigned)K, 3, 3);
    tdm_load_2d((unsigned)(size_t)&Bs[0][0][0], Wbase, 32, 64, (unsigned)K, 3, 3);
  }
#endif

  int buf = 0;
  for (int k0 = 0; k0 < K; k0 += 32, buf ^= 1) {
#if TPI_HAVE_TDM
    if (wave == 0) __builtin_amdgcn_s_wait_tensorcnt(0);
    __syncthreads();                       // tile `buf` visible to all waves
    if (wave == 0 && k0 + 32 < K) {        // prefetch next tile into other buffer
      tdm_load_2d((unsigned)(size_t)&As[buf ^ 1][0][0], Abase + k0 + 32,
                  32, 64, (unsigned)K, 3, 3);
      tdm_load_2d((unsigned)(size_t)&Bs[buf ^ 1][0][0], Wbase + k0 + 32,
                  32, 64, (unsigned)K, 3, 3);
    }
#else
    __syncthreads();
    #pragma unroll
    for (int c = 0; c < 2; ++c) {
      int chunk = tid + c * 128;           // 0..255, 8-half (16B) chunks
      int row = chunk >> 2;
      int cc  = (chunk & 3) * 8;
      *(uint4*)&As[buf][row][cc] = *(const uint4*)&Abase[(size_t)row * K + k0 + cc];
      *(uint4*)&Bs[buf][row][cc] = *(const uint4*)&Wbase[(size_t)row * K + k0 + cc];
    }
    __syncthreads();
#endif

    v16h af = frag_a(&As[buf][wave * 16][0], 40, r, hi);
    #pragma unroll
    for (int g = 0; g < 4; ++g) {
      v16h bf = frag_b(&Bs[buf][g * 16][0], 40, r, hi);
      acc[g] = wmma32(af, bf, acc[g]);
    }
  }

  // Epilogue: C layout element (reg v, lane) = (M = v + 8*hi, N = r).
  #pragma unroll
  for (int g = 0; g < 4; ++g) {
    const int col = colBase + g * 16 + r;
    const float bv = bias[col];
    #pragma unroll
    for (int v = 0; v < 8; ++v) {
      const int row = rowBase + wave * 16 + v + 8 * hi;
      float cv = acc[g][v] + bv;
      if constexpr (RELU) cv = fmaxf(cv, 0.0f);
      const size_t idx = (size_t)row * N + col;
      if constexpr (OUTF32) Cf[idx] = cv;
      if constexpr (OUTF16) Ch[idx] = (_Float16)cv;
    }
  }
}

// ---------------------------------------------------------------------------
// Distance attention (flash-style online softmax), one block per (b,h,64 q rows).
// scores = -sqrt(max(q2 + k2 - 2 q.k, 0)) / sqrt(HD)
// Q/K/V tiles (64 tokens x 64 halves, row stride B*3E) are TDM-loaded.
// ---------------------------------------------------------------------------
__global__ __launch_bounds__(128)
void tpi_attn_kernel(const _Float16* __restrict__ qkv,   // [Ntok, 3E] f16
                     float* __restrict__ out) {          // [Ntok, E] f32
  const int bh    = blockIdx.y;
  const int b     = bh >> 3;          // H == 8
  const int h     = bh & 7;
  const int sBase = blockIdx.x * 64;

  const int tid  = threadIdx.x;
  const int wave = tid >> 5;
  const int lane = tid & 31;
  const int r  = lane & 15;
  const int hi = lane >> 4;

  __shared__ _Float16 Qs[64][72];
  __shared__ _Float16 Ks[64][72];
  __shared__ _Float16 Vs[64][72];
  __shared__ float    q2s[64];
  __shared__ float    k2s[64];
  __shared__ _Float16 Ps[4][16][72];   // per-wave probability tile

  const size_t stride = (size_t)E3;
  const unsigned rowStride = (unsigned)(Bdim * E3);   // halves between seq rows
  const _Float16* qb  = qkv + (size_t)h * HD;
  const _Float16* kb_ = qkv + (size_t)(Edim + h * HD);
  const _Float16* vb  = qkv + (size_t)(2 * Edim + h * HD);

  // Load Q tile (64 seq rows x 64 halves).
#if TPI_HAVE_TDM
  if (wave == 0) {
    tdm_load_2d((unsigned)(size_t)&Qs[0][0],
                qb + (size_t)(sBase * Bdim + b) * stride, 64, 64, rowStride, 4, 3);
    __builtin_amdgcn_s_wait_tensorcnt(0);
  }
  __syncthreads();
#else
  #pragma unroll
  for (int c = 0; c < 4; ++c) {
    int chunk = tid + c * 128;          // 0..511
    int row = chunk >> 3;
    int cc  = (chunk & 7) * 8;
    int tok = (sBase + row) * Bdim + b;
    *(uint4*)&Qs[row][cc] = *(const uint4*)&qb[(size_t)tok * stride + cc];
  }
  __syncthreads();
#endif
  if (tid < 64) {
    float s2 = 0.0f;
    #pragma unroll 8
    for (int d = 0; d < HD; ++d) { float x = (float)Qs[tid][d]; s2 += x * x; }
    q2s[tid] = s2;
  }
  __syncthreads();

  const v16h aq0 = frag_a(&Qs[wave * 16][0],  72, r, hi);
  const v16h aq1 = frag_a(&Qs[wave * 16][32], 72, r, hi);

  float Mrow[8], Srow[8];
  v8f O[4] = {};
  #pragma unroll
  for (int v = 0; v < 8; ++v) { Mrow[v] = -1e30f; Srow[v] = 0.0f; }

  const float inv_s = 0.125f;   // 1/sqrt(64)

  for (int kt = 0; kt < Sdim; kt += 64) {
    __syncthreads();                       // previous tile fully consumed
#if TPI_HAVE_TDM
    if (wave == 0) {
      tdm_load_2d((unsigned)(size_t)&Ks[0][0],
                  kb_ + (size_t)(kt * Bdim + b) * stride, 64, 64, rowStride, 4, 3);
      tdm_load_2d((unsigned)(size_t)&Vs[0][0],
                  vb + (size_t)(kt * Bdim + b) * stride, 64, 64, rowStride, 4, 3);
      __builtin_amdgcn_s_wait_tensorcnt(0);
    }
    __syncthreads();
#else
    #pragma unroll
    for (int c = 0; c < 4; ++c) {
      int chunk = tid + c * 128;
      int row = chunk >> 3;
      int cc  = (chunk & 7) * 8;
      int tok = (kt + row) * Bdim + b;
      *(uint4*)&Ks[row][cc] = *(const uint4*)&kb_[(size_t)tok * stride + cc];
      *(uint4*)&Vs[row][cc] = *(const uint4*)&vb[(size_t)tok * stride + cc];
    }
    __syncthreads();
#endif
    if (tid < 64) {
      float s2 = 0.0f;
      #pragma unroll 8
      for (int d = 0; d < HD; ++d) { float x = (float)Ks[tid][d]; s2 += x * x; }
      k2s[tid] = s2;
    }
    __syncthreads();

    // dot = Q (16x64) . K^T (64 keys); K-dim = HD = 64 -> 2 WMMA steps per group
    v8f dacc[4] = {};
    #pragma unroll
    for (int g = 0; g < 4; ++g) {
      v16h bk0 = frag_b(&Ks[g * 16][0],  72, r, hi);
      dacc[g] = wmma32(aq0, bk0, dacc[g]);
      v16h bk1 = frag_b(&Ks[g * 16][32], 72, r, hi);
      dacc[g] = wmma32(aq1, bk1, dacc[g]);
    }

    // scores
    float sc[4][8];
    #pragma unroll
    for (int g = 0; g < 4; ++g) {
      const float kk = k2s[g * 16 + r];
      #pragma unroll
      for (int v = 0; v < 8; ++v) {
        const int m = wave * 16 + v + 8 * hi;
        float dist = q2s[m] + kk - 2.0f * dacc[g][v];
        sc[g][v] = -sqrtf(fmaxf(dist, 0.0f)) * inv_s;
      }
    }

    // online softmax per row (rows live across 16 lanes with fixed hi bit)
    #pragma unroll
    for (int v = 0; v < 8; ++v) {
      float lm = fmaxf(fmaxf(sc[0][v], sc[1][v]), fmaxf(sc[2][v], sc[3][v]));
      lm = fmaxf(lm, __shfl_xor(lm, 1));
      lm = fmaxf(lm, __shfl_xor(lm, 2));
      lm = fmaxf(lm, __shfl_xor(lm, 4));
      lm = fmaxf(lm, __shfl_xor(lm, 8));
      const float newM  = fmaxf(Mrow[v], lm);
      const float scale = __expf(Mrow[v] - newM);
      float rs = 0.0f;
      #pragma unroll
      for (int g = 0; g < 4; ++g) {
        float p = __expf(sc[g][v] - newM);
        sc[g][v] = p;
        rs += p;
      }
      rs += __shfl_xor(rs, 1);
      rs += __shfl_xor(rs, 2);
      rs += __shfl_xor(rs, 4);
      rs += __shfl_xor(rs, 8);
      Srow[v] = Srow[v] * scale + rs;
      Mrow[v] = newM;
      #pragma unroll
      for (int g = 0; g < 4; ++g) O[g][v] *= scale;
    }

    // Store P (f16) into per-wave LDS, re-gather as A fragments.
    #pragma unroll
    for (int g = 0; g < 4; ++g) {
      #pragma unroll
      for (int v = 0; v < 8; ++v)
        Ps[wave][v + 8 * hi][g * 16 + r] = (_Float16)sc[g][v];
    }
    asm volatile("s_wait_dscnt 0" ::: "memory");   // same-wave LDS RAW

    v16h ap0 = frag_a(&Ps[wave][0][0],  72, r, hi);
    v16h ap1 = frag_a(&Ps[wave][0][32], 72, r, hi);

    // O += P (16x64) . V (64 keys x 64 hd)
    #pragma unroll
    for (int g = 0; g < 4; ++g) {
      v16h bv0 = frag_bT(&Vs[0][g * 16],  72, r, hi);
      O[g] = wmma32(ap0, bv0, O[g]);
      v16h bv1 = frag_bT(&Vs[32][g * 16], 72, r, hi);
      O[g] = wmma32(ap1, bv1, O[g]);
    }
  }

  // Normalize and write out[s,b, h*64 + d]
  #pragma unroll
  for (int g = 0; g < 4; ++g) {
    #pragma unroll
    for (int v = 0; v < 8; ++v) {
      const int srow = sBase + wave * 16 + v + 8 * hi;
      const int tok  = srow * Bdim + b;
      out[(size_t)tok * Edim + h * HD + g * 16 + r] = O[g][v] / Srow[v];
    }
  }
}

// ---------------------------------------------------------------------------
// y = LayerNorm(x + res) * g + b ; also emit f16 shadow. One block per token.
// ---------------------------------------------------------------------------
__global__ __launch_bounds__(256)
void tpi_add_ln_kernel(const float* __restrict__ x,
                       const float* __restrict__ res,
                       const float* __restrict__ gw,
                       const float* __restrict__ bw,
                       float* __restrict__ y,
                       _Float16* __restrict__ yh) {
  const int t   = blockIdx.x;
  const int tid = threadIdx.x;
  const size_t base = (size_t)t * Edim;

  float v0 = x[base + tid]       + res[base + tid];
  float v1 = x[base + tid + 256] + res[base + tid + 256];

  __shared__ float part[8];
  float s = v0 + v1;
  s += __shfl_xor(s, 1);  s += __shfl_xor(s, 2);
  s += __shfl_xor(s, 4);  s += __shfl_xor(s, 8);
  s += __shfl_xor(s, 16);
  if ((tid & 31) == 0) part[tid >> 5] = s;
  __syncthreads();
  float tot = 0.0f;
  #pragma unroll
  for (int i = 0; i < 8; ++i) tot += part[i];
  const float mean = tot * (1.0f / Edim);

  const float d0 = v0 - mean, d1 = v1 - mean;
  float q = d0 * d0 + d1 * d1;
  q += __shfl_xor(q, 1);  q += __shfl_xor(q, 2);
  q += __shfl_xor(q, 4);  q += __shfl_xor(q, 8);
  q += __shfl_xor(q, 16);
  __syncthreads();
  if ((tid & 31) == 0) part[tid >> 5] = q;
  __syncthreads();
  float qtot = 0.0f;
  #pragma unroll
  for (int i = 0; i < 8; ++i) qtot += part[i];
  const float rstd = rsqrtf(qtot * (1.0f / Edim) + 1e-5f);

  const float o0 = d0 * rstd * gw[tid]       + bw[tid];
  const float o1 = d1 * rstd * gw[tid + 256] + bw[tid + 256];
  y[base + tid]        = o0;
  y[base + tid + 256]  = o1;
  yh[base + tid]       = (_Float16)o0;
  yh[base + tid + 256] = (_Float16)o1;
}

// ---------------------------------------------------------------------------
// Host-side orchestration
// ---------------------------------------------------------------------------
static inline int grid1(int n, int b) { return (n + b - 1) / b; }

extern "C" void kernel_launch(void* const* d_in, const int* in_sizes, int n_in,
                              void* d_out, int out_size, void* d_ws, size_t ws_size,
                              hipStream_t stream) {
  (void)in_sizes; (void)n_in; (void)out_size; (void)ws_size;

  const float* x_in = (const float*)d_in[0];
  const float* ipw  = (const float*)d_in[1];
  const float* ipb  = (const float*)d_in[2];
  const float* opw  = (const float*)d_in[3];
  const float* opb  = (const float*)d_in[4];
  const float* w1   = (const float*)d_in[5];
  const float* b1   = (const float*)d_in[6];
  const float* w2   = (const float*)d_in[7];
  const float* b2   = (const float*)d_in[8];
  const float* g1   = (const float*)d_in[9];
  const float* be1  = (const float*)d_in[10];
  const float* g2   = (const float*)d_in[11];
  const float* be2  = (const float*)d_in[12];

  // Workspace arena (~134 MB)
  char* ws = (char*)d_ws;
  size_t off = 0;
  auto carve = [&](size_t bytes) -> char* {
    char* p = ws + off;
    off = (off + bytes + 255) & ~(size_t)255;
    return p;
  };
  float*    xbuf   = (float*)   carve((size_t)Ntok * Edim * 4);
  _Float16* x_h    = (_Float16*)carve((size_t)Ntok * Edim * 2);
  _Float16* qkv_h  = (_Float16*)carve((size_t)Ntok * E3   * 2);
  _Float16* attn_h = (_Float16*)carve((size_t)Ntok * Edim * 2);
  float*    buf_a  = (float*)   carve((size_t)Ntok * Edim * 4);
  _Float16* ff1_h  = (_Float16*)carve((size_t)Ntok * FFd  * 2);
  _Float16* ipw_h  = (_Float16*)carve((size_t)Ldim * E3  * Edim * 2);
  _Float16* opw_h  = (_Float16*)carve((size_t)Ldim * Edim * Edim * 2);
  _Float16* w1_h   = (_Float16*)carve((size_t)Ldim * FFd * Edim * 2);
  _Float16* w2_h   = (_Float16*)carve((size_t)Ldim * Edim * FFd * 2);

  const int NE = Ntok * Edim;

  // One-time (per call) weight conversion to f16.
  tpi_cvt_f16_kernel<<<grid1(Ldim * E3 * Edim, 256), 256, 0, stream>>>(ipw, ipw_h, Ldim * E3 * Edim);
  tpi_cvt_f16_kernel<<<grid1(Ldim * Edim * Edim, 256), 256, 0, stream>>>(opw, opw_h, Ldim * Edim * Edim);
  tpi_cvt_f16_kernel<<<grid1(Ldim * FFd * Edim, 256), 256, 0, stream>>>(w1, w1_h, Ldim * FFd * Edim);
  tpi_cvt_f16_kernel<<<grid1(Ldim * Edim * FFd, 256), 256, 0, stream>>>(w2, w2_h, Ldim * Edim * FFd);

  // x -> xbuf (f32 working copy) and f16 shadow.
  tpi_copy_f32_kernel<<<grid1(NE, 256), 256, 0, stream>>>(x_in, xbuf, NE);
  tpi_cvt_f16_kernel<<<grid1(NE, 256), 256, 0, stream>>>(xbuf, x_h, NE);

  const dim3 blkG(128);
  for (int i = 0; i < Ldim; ++i) {
    // 1) QKV: [N,512] x [1536,512]^T -> f16
    tpi_gemm_kernel<false, false, true>
        <<<dim3(E3 / 64, Ntok / 64), blkG, 0, stream>>>(
            x_h, ipw_h + (size_t)i * E3 * Edim, ipb + (size_t)i * E3,
            nullptr, qkv_h, Ntok, E3, Edim);

    // 2) Distance attention -> buf_a (f32)
    tpi_attn_kernel<<<dim3(Sdim / 64, Bdim * Hh), blkG, 0, stream>>>(qkv_h, buf_a);

    // 3) f16 shadow of attention output
    tpi_cvt_f16_kernel<<<grid1(NE, 256), 256, 0, stream>>>(buf_a, attn_h, NE);

    // 4) Output projection -> buf_a (f32)
    tpi_gemm_kernel<false, true, false>
        <<<dim3(Edim / 64, Ntok / 64), blkG, 0, stream>>>(
            attn_h, opw_h + (size_t)i * Edim * Edim, opb + (size_t)i * Edim,
            buf_a, nullptr, Ntok, Edim, Edim);

    // 5) x = LN(x + attn)
    tpi_add_ln_kernel<<<Ntok, 256, 0, stream>>>(
        xbuf, buf_a, g1 + (size_t)i * Edim, be1 + (size_t)i * Edim, xbuf, x_h);

    // 6) FFN up + ReLU -> f16
    tpi_gemm_kernel<true, false, true>
        <<<dim3(FFd / 64, Ntok / 64), blkG, 0, stream>>>(
            x_h, w1_h + (size_t)i * FFd * Edim, b1 + (size_t)i * FFd,
            nullptr, ff1_h, Ntok, FFd, Edim);

    // 7) FFN down -> buf_a (f32)
    tpi_gemm_kernel<false, true, false>
        <<<dim3(Edim / 64, Ntok / 64), blkG, 0, stream>>>(
            ff1_h, w2_h + (size_t)i * Edim * FFd, b2 + (size_t)i * Edim,
            buf_a, nullptr, Ntok, Edim, FFd);

    // 8) x = LN(x + ff)
    tpi_add_ln_kernel<<<Ntok, 256, 0, stream>>>(
        xbuf, buf_a, g2 + (size_t)i * Edim, be2 + (size_t)i * Edim, xbuf, x_h);
  }

  tpi_copy_f32_kernel<<<grid1(NE, 256), 256, 0, stream>>>(xbuf, (float*)d_out, NE);
}
